// DVBF_27891517620863
// MI455X (gfx1250) — matrix operand order repaired and checked by reading, not compiled, not run on hardware
//
#include <hip/hip_runtime.h>

#define Tn     64
#define Bsz    2048
#define LAT    128
#define CTRLD  16
#define BM     16
#define NTH    256

typedef __attribute__((ext_vector_type(16))) __bf16 v16bf;
typedef __attribute__((ext_vector_type(8)))  float  v8f;

__device__ __forceinline__ __bf16 f2bf(float x) {
  union { float f; unsigned u; } in; in.f = x;
  unsigned r = in.u + 0x7FFFu + ((in.u >> 16) & 1u);   // round-to-nearest-even
  union { unsigned short s; __bf16 b; } out;
  out.s = (unsigned short)(r >> 16);
  return out.b;
}

__device__ __forceinline__ float frcp(float x) {
#if __has_builtin(__builtin_amdgcn_rcpf)
  return __builtin_amdgcn_rcpf(x);
#else
  return 1.0f / x;
#endif
}

__device__ __forceinline__ float ftanh(float x) {
#if __has_builtin(__builtin_amdgcn_tanhf)
  return __builtin_amdgcn_tanhf(x);
#elif __has_builtin(__builtin_amdgcn_tanh_f32)
  return __builtin_amdgcn_tanh_f32(x);
#else
  return tanhf(x);
#endif
}

__device__ __forceinline__ float sigf(float x) { return frcp(1.0f + __expf(-x)); }

// ---- WMMA fragment helpers (contiguous k = 16*g + s layouts, wave32) ------------------
// A-fragment K-offset pattern for element pair v (lane half g)
__device__ __forceinline__ int a_koff(int v, int g) {
  return 2 * v + 16 * g;
}

// A (16x32 bf16) from LDS bf16 staging: contiguous K pairs -> ds_load_b128 x2
__device__ __forceinline__ v16bf loadA_lds(const __bf16 (*Sa)[320], int kt, int lane) {
  const int m = lane & 15, g = lane >> 4;
  v16bf a;
#pragma unroll
  for (int v = 0; v < 8; ++v) {
    const int kb = kt * 32 + a_koff(v, g);
    a[2 * v]     = Sa[m][kb];
    a[2 * v + 1] = Sa[m][kb + 1];
  }
  return a;
}

// A (16x32 bf16) built from f32 LDS row-array (row stride vld), K window kt*32..+31
__device__ __forceinline__ v16bf buildA_f32(const float* __restrict__ vec, int vld,
                                            int kt, int lane) {
  const int m = lane & 15, g = lane >> 4;
  v16bf a;
#pragma unroll
  for (int v = 0; v < 8; ++v) {
    const int kb = kt * 32 + a_koff(v, g);
    a[2 * v]     = f2bf(vec[m * vld + kb]);
    a[2 * v + 1] = f2bf(vec[m * vld + kb + 1]);
  }
  return a;
}

// A (16x32 bf16) from f32 LDS with only K<16 valid (u operand, zero-padded)
__device__ __forceinline__ v16bf buildA_pad16(const float* __restrict__ vec, int vld,
                                              int lane) {
  const int m = lane & 15, g = lane >> 4;
  v16bf a;
#pragma unroll
  for (int v = 0; v < 8; ++v) {
    const int kb = a_koff(v, g);
    a[2 * v]     = (kb < 16)     ? f2bf(vec[m * vld + kb])     : f2bf(0.0f);
    a[2 * v + 1] = (kb + 1 < 16) ? f2bf(vec[m * vld + kb + 1]) : f2bf(0.0f);
  }
  return a;
}

// B (32x16 bf16): weights pre-shuffled so each lane's 16 values are contiguous (32B load)
__device__ __forceinline__ v16bf loadB_frag(const __bf16* __restrict__ Wf, int ktiles,
                                            int kt, int nt, int lane) {
  return *reinterpret_cast<const v16bf*>(Wf + (((size_t)nt * ktiles + kt) * 32 + lane) * 16);
}

// Dense GEMM: out(16xN,f32,ld=ldo) = Sa(16xK,bf16) @ Wf(frag layout) + bias
__device__ __forceinline__ void gemm(const __bf16 (*Sa)[320], const __bf16* __restrict__ Wf,
                                     const float* __restrict__ bias, float* __restrict__ out,
                                     int ldo, int K, int N, int lane, int wid) {
  const int ktiles = K >> 5, ntiles = N >> 4;
  for (int t = wid; t < ntiles; t += 8) {
    v8f acc;
#pragma unroll
    for (int r = 0; r < 8; ++r) acc[r] = 0.0f;
    for (int kt = 0; kt < ktiles; ++kt) {
      v16bf a = loadA_lds(Sa, kt, lane);
      v16bf b = loadB_frag(Wf, ktiles, kt, t, lane);
      acc = __builtin_amdgcn_wmma_f32_16x16x32_bf16(false, a, false, b, (short)0, acc,
                                                    false, false);
    }
    const int n = t * 16 + (lane & 15), g = lane >> 4;
    const float bv = bias ? bias[n] : 0.0f;
#pragma unroll
    for (int r = 0; r < 8; ++r) out[(size_t)(r + 8 * g) * ldo + n] = acc[r] + bv;
  }
}

// ---- weight prep: f32 -> bf16 fragment-ready layout ----------------------
__global__ void k_frag_dense(const float* __restrict__ src, __bf16* __restrict__ dst,
                             int K, int N, int Kvalid) {
  int idx = blockIdx.x * blockDim.x + threadIdx.x;
  if (idx >= K * N) return;
  const int e = idx & 15, lane = (idx >> 4) & 31, blk = idx >> 9;
  const int ktiles = K >> 5;
  const int kt = blk % ktiles, nt = blk / ktiles;
  const int n = nt * 16 + (lane & 15);
  const int k = kt * 32 + (lane >> 4) * 16 + e;
  dst[idx] = (k < Kvalid) ? f2bf(src[(size_t)k * N + n]) : f2bf(0.0f);
}

// LSTM concat: rows 0..127 from Wx, 128..255 from Wh  (K=256, N=512)
__global__ void k_frag_lstm(const float* __restrict__ Wx, const float* __restrict__ Wh,
                            __bf16* __restrict__ dst) {
  int idx = blockIdx.x * blockDim.x + threadIdx.x;
  if (idx >= 256 * 512) return;
  const int e = idx & 15, lane = (idx >> 4) & 31, blk = idx >> 9;
  const int kt = blk % 8, nt = blk / 8;
  const int n = nt * 16 + (lane & 15);
  const int k = kt * 32 + (lane >> 4) * 16 + e;
  const float v = (k < 128) ? Wx[(size_t)k * 512 + n] : Wh[(size_t)(k - 128) * 512 + n];
  dst[idx] = f2bf(v);
}

// Bank transpose: logical Wbig[k][n], n = im*128+j, value src[n*Kb + k] (src[im][j][k]).
// K possibly padded beyond Kvalid with zeros. N = 2048.
__global__ void k_frag_bankT(const float* __restrict__ src, __bf16* __restrict__ dst,
                             int Kb, int K, int Kvalid) {
  int idx = blockIdx.x * blockDim.x + threadIdx.x;
  if (idx >= K * 2048) return;
  const int e = idx & 15, lane = (idx >> 4) & 31, blk = idx >> 9;
  const int ktiles = K >> 5;
  const int kt = blk % ktiles, nt = blk / ktiles;
  const int n = nt * 16 + (lane & 15);
  const int k = kt * 32 + (lane >> 4) * 16 + e;
  dst[idx] = (k < Kvalid) ? f2bf(src[(size_t)n * Kb + k]) : f2bf(0.0f);
}

// ---- main persistent kernel: one block = 16 batch rows, all T steps ------
__global__ __launch_bounds__(NTH, 2) void dvbf_main(
    const float* __restrict__ xs, const float* __restrict__ us,
    const float* __restrict__ eps1, const float* __restrict__ eps,
    const float* __restrict__ lstm_b, const float* __restrict__ init_b1,
    const float* __restrict__ init_b2, const float* __restrict__ trans_b1,
    const float* __restrict__ trans_b2, const float* __restrict__ obs_b1,
    const float* __restrict__ obs_b2, const float* __restrict__ rec_b1,
    const float* __restrict__ rec_b2,
    const __bf16* __restrict__ W_lstm, const __bf16* __restrict__ W_init1,
    const __bf16* __restrict__ W_init2, const __bf16* __restrict__ W_trans1,
    const __bf16* __restrict__ W_trans2, const __bf16* __restrict__ W_obs1,
    const __bf16* __restrict__ W_obs2, const __bf16* __restrict__ W_rec1,
    const __bf16* __restrict__ W_rec2, const __bf16* __restrict__ W_alpha,
    const __bf16* __restrict__ W_A, const __bf16* __restrict__ W_Bm,
    const __bf16* __restrict__ W_C,
    float* __restrict__ out_zs, float* __restrict__ out_xr) {
  __shared__ float  Sg[BM][512];     // GEMM outputs (gates / dense results)
  __shared__ __bf16 Sa[BM][320];     // bf16 A-operand staging
  __shared__ float  Szh[BM][128];    // h (LSTM phase) / z (scan phase)
  __shared__ float  Swc[BM][128];    // c (LSTM phase) / w (scan phase)
  __shared__ float  Salpha[BM][16];
  __shared__ float  Su[BM][16];

  const int tid  = threadIdx.x;
  const int lane = tid & 31;
  const int wid  = tid >> 5;
  const int b0   = blockIdx.x * BM;

  // ---------------- Phase A: LSTM over T ----------------
  for (int i = tid; i < BM * 128; i += NTH) {
    Szh[i >> 7][i & 127] = 0.0f;
    Swc[i >> 7][i & 127] = 0.0f;
  }
  __syncthreads();

  for (int t = 0; t < Tn; ++t) {
    for (int i = tid; i < BM * 32; i += NTH) {
      const int m = i >> 5, k4 = (i & 31) * 4;
      const float4 x = *(const float4*)&xs[((size_t)t * Bsz + b0 + m) * 128 + k4];
      Sa[m][k4 + 0] = f2bf(x.x); Sa[m][k4 + 1] = f2bf(x.y);
      Sa[m][k4 + 2] = f2bf(x.z); Sa[m][k4 + 3] = f2bf(x.w);
      const float4 h = *(const float4*)&Szh[m][k4];
      Sa[m][128 + k4 + 0] = f2bf(h.x); Sa[m][128 + k4 + 1] = f2bf(h.y);
      Sa[m][128 + k4 + 2] = f2bf(h.z); Sa[m][128 + k4 + 3] = f2bf(h.w);
    }
    if (t + 1 < Tn)
      __builtin_prefetch(&xs[((size_t)(t + 1) * Bsz + b0 + (tid & 15)) * 128], 0, 1);
    __syncthreads();
    gemm(Sa, W_lstm, lstm_b, &Sg[0][0], 512, 256, 512, lane, wid);
    __syncthreads();
    for (int i = tid; i < BM * 128; i += NTH) {
      const int m = i >> 7, j = i & 127;
      const float ig = sigf(Sg[m][j]);
      const float fg = sigf(Sg[m][128 + j]);
      const float gg = ftanh(Sg[m][256 + j]);
      const float og = sigf(Sg[m][384 + j]);
      const float c = fg * Swc[m][j] + ig * gg;
      Swc[m][j] = c;
      Szh[m][j] = og * ftanh(c);
    }
    __syncthreads();
  }

  // ---------------- Phase B: initial network + initial transition ----------
  for (int i = tid; i < BM * 128; i += NTH) Sa[i >> 7][i & 127] = f2bf(Szh[i >> 7][i & 127]);
  __syncthreads();
  gemm(Sa, W_init1, init_b1, &Sg[0][0], 512, 128, 128, lane, wid);
  __syncthreads();
  for (int i = tid; i < BM * 128; i += NTH)
    Sa[i >> 7][i & 127] = f2bf(fmaxf(Sg[i >> 7][i & 127], 0.0f));
  __syncthreads();
  gemm(Sa, W_init2, init_b2, &Sg[0][0], 512, 128, 256, lane, wid);
  __syncthreads();
  for (int i = tid; i < BM * 32; i += NTH) {
    const int m = i >> 5, k4 = (i & 31) * 4;
    const float4 ep = *(const float4*)&eps1[((size_t)(b0 + m)) * 128 + k4];
    Sa[m][k4 + 0] = f2bf(Sg[m][k4 + 0] + __expf(0.5f * Sg[m][128 + k4 + 0]) * ep.x);
    Sa[m][k4 + 1] = f2bf(Sg[m][k4 + 1] + __expf(0.5f * Sg[m][128 + k4 + 1]) * ep.y);
    Sa[m][k4 + 2] = f2bf(Sg[m][k4 + 2] + __expf(0.5f * Sg[m][128 + k4 + 2]) * ep.z);
    Sa[m][k4 + 3] = f2bf(Sg[m][k4 + 3] + __expf(0.5f * Sg[m][128 + k4 + 3]) * ep.w);
  }
  __syncthreads();
  gemm(Sa, W_trans1, trans_b1, &Sg[0][0], 512, 128, 128, lane, wid);
  __syncthreads();
  for (int i = tid; i < BM * 128; i += NTH)
    Sa[i >> 7][i & 127] = f2bf(fmaxf(Sg[i >> 7][i & 127], 0.0f));
  __syncthreads();
  gemm(Sa, W_trans2, trans_b2, &Sg[0][0], 512, 128, 128, lane, wid);
  __syncthreads();
  for (int i = tid; i < BM * 32; i += NTH) {
    const int m = i >> 5, k4 = (i & 31) * 4;
    const float4 zv = *(const float4*)&Sg[m][k4];
    *(float4*)&Szh[m][k4] = zv;
    *(float4*)&out_zs[((size_t)(b0 + m)) * 128 + k4] = zv;    // zs row t=0
  }
  __syncthreads();

  // ---------------- Phase C: DVBF scan over t = 0..T-2 ----------------
  for (int t = 0; t < Tn - 1; ++t) {
    // stage u_t (f32), and Sa = [bf16(z) | bf16(u) | 0-pad] (K=160 alpha, K=128 obs1)
    for (int i = tid; i < BM * 4; i += NTH) {
      const int m = i >> 2, k4 = (i & 3) * 4;
      *(float4*)&Su[m][k4] = *(const float4*)&us[((size_t)t * Bsz + b0 + m) * CTRLD + k4];
    }
    for (int i = tid; i < BM * 32; i += NTH) {
      const int m = i >> 5, k4 = (i & 31) * 4;
      const float4 zv = *(const float4*)&Szh[m][k4];
      Sa[m][k4 + 0] = f2bf(zv.x); Sa[m][k4 + 1] = f2bf(zv.y);
      Sa[m][k4 + 2] = f2bf(zv.z); Sa[m][k4 + 3] = f2bf(zv.w);
    }
    for (int i = tid; i < BM * 32; i += NTH) {
      const int m = i >> 5, k = i & 31;
      Sa[m][128 + k] = (k < CTRLD)
          ? f2bf(us[((size_t)t * Bsz + b0 + m) * CTRLD + k]) : f2bf(0.0f);
    }
    if (t + 2 < Tn)
      __builtin_prefetch(&xs[((size_t)(t + 2) * Bsz + b0 + (tid & 15)) * 128], 0, 1);
    __syncthreads();

    // alpha logits (WMMA, 1 tile) + obs layer 1
    gemm(Sa, W_alpha, nullptr, &Salpha[0][0], 16, 160, 16, lane, wid);
    gemm(Sa, W_obs1, obs_b1, &Sg[0][0], 512, 128, 128, lane, wid);
    __syncthreads();

    // relu(obs1) -> Sa ; softmax read phase in registers
    for (int i = tid; i < BM * 128; i += NTH)
      Sa[i >> 7][i & 127] = f2bf(fmaxf(Sg[i >> 7][i & 127], 0.0f));
    float alpha_v;
    {
      const int m = tid >> 4, i = tid & 15;
      float mx = Salpha[m][0];
      for (int j = 1; j < 16; ++j) mx = fmaxf(mx, Salpha[m][j]);
      float den = 0.0f;
      for (int j = 0; j < 16; ++j) den += __expf(Salpha[m][j] - mx);
      alpha_v = __expf(Salpha[m][i] - mx) * frcp(den);
    }
    __syncthreads();
    { const int m = tid >> 4, i = tid & 15; Salpha[m][i] = alpha_v; }
    gemm(Sa, W_obs2, obs_b2, &Sg[0][0], 512, 128, 128, lane, wid);
    __syncthreads();

    // emit x_mean; stage rec concat [z | x_{t+1} | u | pad] (K=288)
    for (int i = tid; i < BM * 32; i += NTH) {
      const int m = i >> 5, k4 = (i & 31) * 4;
      *(float4*)&out_xr[((size_t)t * Bsz + b0 + m) * 128 + k4] = *(const float4*)&Sg[m][k4];
      const float4 zv = *(const float4*)&Szh[m][k4];
      Sa[m][k4 + 0] = f2bf(zv.x); Sa[m][k4 + 1] = f2bf(zv.y);
      Sa[m][k4 + 2] = f2bf(zv.z); Sa[m][k4 + 3] = f2bf(zv.w);
      const float4 xn = *(const float4*)&xs[((size_t)(t + 1) * Bsz + b0 + m) * 128 + k4];
      Sa[m][128 + k4 + 0] = f2bf(xn.x); Sa[m][128 + k4 + 1] = f2bf(xn.y);
      Sa[m][128 + k4 + 2] = f2bf(xn.z); Sa[m][128 + k4 + 3] = f2bf(xn.w);
    }
    for (int i = tid; i < BM * 32; i += NTH) {
      const int m = i >> 5, k = i & 31;
      Sa[m][256 + k] = (k < CTRLD) ? f2bf(Su[m][k]) : f2bf(0.0f);
    }
    __syncthreads();

    gemm(Sa, W_rec1, rec_b1, &Sg[0][0], 512, 288, 128, lane, wid);
    __syncthreads();
    for (int i = tid; i < BM * 128; i += NTH)
      Sa[i >> 7][i & 127] = f2bf(fmaxf(Sg[i >> 7][i & 127], 0.0f));
    __syncthreads();
    gemm(Sa, W_rec2, rec_b2, &Sg[0][0], 512, 128, 256, lane, wid);
    __syncthreads();

    // sample w -> Swc
    for (int i = tid; i < BM * 32; i += NTH) {
      const int m = i >> 5, k4 = (i & 31) * 4;
      const float4 ep = *(const float4*)&eps[((size_t)t * Bsz + b0 + m) * 128 + k4];
      Swc[m][k4 + 0] = Sg[m][k4 + 0] + __expf(0.5f * Sg[m][128 + k4 + 0]) * ep.x;
      Swc[m][k4 + 1] = Sg[m][k4 + 1] + __expf(0.5f * Sg[m][128 + k4 + 1]) * ep.y;
      Swc[m][k4 + 2] = Sg[m][k4 + 2] + __expf(0.5f * Sg[m][128 + k4 + 2]) * ep.z;
      Swc[m][k4 + 3] = Sg[m][k4 + 3] + __expf(0.5f * Sg[m][128 + k4 + 3]) * ep.w;
    }
    __syncthreads();

    // z_next[m][j] = sum_i alpha[m][i] * (A_i@z + B_i@u + C_i@w)[m][j]
    // Per-mixture partials via 9 chained WMMAs; alpha applied in f32 afterwards.
    {
      v16bf zf0 = buildA_f32(&Szh[0][0], 128, 0, lane);
      v16bf zf1 = buildA_f32(&Szh[0][0], 128, 1, lane);
      v16bf zf2 = buildA_f32(&Szh[0][0], 128, 2, lane);
      v16bf zf3 = buildA_f32(&Szh[0][0], 128, 3, lane);
      v16bf wf0 = buildA_f32(&Swc[0][0], 128, 0, lane);
      v16bf wf1 = buildA_f32(&Swc[0][0], 128, 1, lane);
      v16bf wf2 = buildA_f32(&Swc[0][0], 128, 2, lane);
      v16bf wf3 = buildA_f32(&Swc[0][0], 128, 3, lane);
      v16bf uf  = buildA_pad16(&Su[0][0], 16, lane);
      const int g = lane >> 4;
      v8f acc;
#pragma unroll
      for (int r = 0; r < 8; ++r) acc[r] = 0.0f;
      for (int i = 0; i < 16; ++i) {
        const int nt = i * 8 + wid;
        v8f y;
#pragma unroll
        for (int r = 0; r < 8; ++r) y[r] = 0.0f;
        y = __builtin_amdgcn_wmma_f32_16x16x32_bf16(false, zf0,
              false, loadB_frag(W_A, 4, 0, nt, lane), (short)0, y, false, false);
        y = __builtin_amdgcn_wmma_f32_16x16x32_bf16(false, zf1,
              false, loadB_frag(W_A, 4, 1, nt, lane), (short)0, y, false, false);
        y = __builtin_amdgcn_wmma_f32_16x16x32_bf16(false, zf2,
              false, loadB_frag(W_A, 4, 2, nt, lane), (short)0, y, false, false);
        y = __builtin_amdgcn_wmma_f32_16x16x32_bf16(false, zf3,
              false, loadB_frag(W_A, 4, 3, nt, lane), (short)0, y, false, false);
        y = __builtin_amdgcn_wmma_f32_16x16x32_bf16(false, uf,
              false, loadB_frag(W_Bm, 1, 0, nt, lane), (short)0, y, false, false);
        y = __builtin_amdgcn_wmma_f32_16x16x32_bf16(false, wf0,
              false, loadB_frag(W_C, 4, 0, nt, lane), (short)0, y, false, false);
        y = __builtin_amdgcn_wmma_f32_16x16x32_bf16(false, wf1,
              false, loadB_frag(W_C, 4, 1, nt, lane), (short)0, y, false, false);
        y = __builtin_amdgcn_wmma_f32_16x16x32_bf16(false, wf2,
              false, loadB_frag(W_C, 4, 2, nt, lane), (short)0, y, false, false);
        y = __builtin_amdgcn_wmma_f32_16x16x32_bf16(false, wf3,
              false, loadB_frag(W_C, 4, 3, nt, lane), (short)0, y, false, false);
#pragma unroll
        for (int r = 0; r < 8; ++r) acc[r] += Salpha[r + 8 * g][i] * y[r];
      }
      const int n = wid * 16 + (lane & 15);
#pragma unroll
      for (int r = 0; r < 8; ++r) Sg[r + 8 * g][n] = acc[r];
    }
    __syncthreads();
    for (int i = tid; i < BM * 32; i += NTH) {
      const int m = i >> 5, k4 = (i & 31) * 4;
      const float4 zn = *(const float4*)&Sg[m][k4];
      *(float4*)&Szh[m][k4] = zn;
      *(float4*)&out_zs[((size_t)(t + 1) * Bsz + b0 + m) * 128 + k4] = zn;
    }
    __syncthreads();
  }
}

extern "C" void kernel_launch(void* const* d_in, const int* in_sizes, int n_in,
                              void* d_out, int out_size, void* d_ws, size_t ws_size,
                              hipStream_t stream) {
  const float* xs       = (const float*)d_in[0];
  const float* us       = (const float*)d_in[1];
  const float* eps1     = (const float*)d_in[2];
  const float* eps      = (const float*)d_in[3];
  const float* lstm_Wx  = (const float*)d_in[4];
  const float* lstm_Wh  = (const float*)d_in[5];
  const float* lstm_b   = (const float*)d_in[6];
  const float* init_W1  = (const float*)d_in[7];
  const float* init_b1  = (const float*)d_in[8];
  const float* init_W2  = (const float*)d_in[9];
  const float* init_b2  = (const float*)d_in[10];
  const float* trans_W1 = (const float*)d_in[11];
  const float* trans_b1 = (const float*)d_in[12];
  const float* trans_W2 = (const float*)d_in[13];
  const float* trans_b2 = (const float*)d_in[14];
  const float* obs_W1   = (const float*)d_in[15];
  const float* obs_b1   = (const float*)d_in[16];
  const float* obs_W2   = (const float*)d_in[17];
  const float* obs_b2   = (const float*)d_in[18];
  const float* rec_W1   = (const float*)d_in[19];
  const float* rec_b1   = (const float*)d_in[20];
  const float* rec_W2   = (const float*)d_in[21];
  const float* rec_b2   = (const float*)d_in[22];
  const float* alpha_W  = (const float*)d_in[23];
  const float* A        = (const float*)d_in[24];
  const float* Bm       = (const float*)d_in[25];
  const float* C        = (const float*)d_in[26];

  __bf16* wsb      = (__bf16*)d_ws;
  __bf16* W_lstm   = wsb;                    // 256x512
  __bf16* W_init1  = W_lstm   + 256 * 512;   // 128x128
  __bf16* W_init2  = W_init1  + 128 * 128;   // 128x256
  __bf16* W_trans1 = W_init2  + 128 * 256;
  __bf16* W_trans2 = W_trans1 + 128 * 128;
  __bf16* W_obs1   = W_trans2 + 128 * 128;
  __bf16* W_obs2   = W_obs1   + 128 * 128;
  __bf16* W_rec1   = W_obs2   + 128 * 128;   // 288x128 (K-padded)
  __bf16* W_rec2   = W_rec1   + 288 * 128;   // 128x256
  __bf16* W_alpha  = W_rec2   + 128 * 256;   // 160x16 (K-padded)
  __bf16* W_A      = W_alpha  + 160 * 16;    // 128 x 2048 (transposed bank)
  __bf16* W_Bm     = W_A      + 128 * 2048;  // 32 x 2048 (K-padded 16->32)
  __bf16* W_C      = W_Bm     + 32 * 2048;   // 128 x 2048

  auto blk = [](int n) { return (n + NTH - 1) / NTH; };
  k_frag_lstm<<<blk(256 * 512), NTH, 0, stream>>>(lstm_Wx, lstm_Wh, W_lstm);
  k_frag_dense<<<blk(128 * 128), NTH, 0, stream>>>(init_W1, W_init1, 128, 128, 128);
  k_frag_dense<<<blk(128 * 256), NTH, 0, stream>>>(init_W2, W_init2, 128, 256, 128);
  k_frag_dense<<<blk(128 * 128), NTH, 0, stream>>>(trans_W1, W_trans1, 128, 128, 128);
  k_frag_dense<<<blk(128 * 128), NTH, 0, stream>>>(trans_W2, W_trans2, 128, 128, 128);
  k_frag_dense<<<blk(128 * 128), NTH, 0, stream>>>(obs_W1, W_obs1, 128, 128, 128);
  k_frag_dense<<<blk(128 * 128), NTH, 0, stream>>>(obs_W2, W_obs2, 128, 128, 128);
  k_frag_dense<<<blk(288 * 128), NTH, 0, stream>>>(rec_W1, W_rec1, 288, 128, 272);
  k_frag_dense<<<blk(128 * 256), NTH, 0, stream>>>(rec_W2, W_rec2, 128, 256, 128);
  k_frag_dense<<<blk(160 * 16), NTH, 0, stream>>>(alpha_W, W_alpha, 160, 16, 144);
  k_frag_bankT<<<blk(128 * 2048), NTH, 0, stream>>>(A,  W_A,  128, 128, 128);
  k_frag_bankT<<<blk(32 * 2048),  NTH, 0, stream>>>(Bm, W_Bm, 16, 32, 16);
  k_frag_bankT<<<blk(128 * 2048), NTH, 0, stream>>>(C,  W_C,  128, 128, 128);

  float* out_zs = (float*)d_out;
  float* out_xr = out_zs + (size_t)Tn * Bsz * 128;

  dvbf_main<<<dim3(Bsz / BM), dim3(NTH), 0, stream>>>(
      xs, us, eps1, eps, lstm_b, init_b1, init_b2, trans_b1, trans_b2, obs_b1, obs_b2,
      rec_b1, rec_b2, W_lstm, W_init1, W_init2, W_trans1, W_trans2, W_obs1, W_obs2,
      W_rec1, W_rec2, W_alpha, W_A, W_Bm, W_C, out_zs, out_xr);
}
